// CutlassGptOssExperts_85598698209591
// MI455X (gfx1250) — compile-verified, matrix operand order checked by reading, not run on hardware
//
#include <hip/hip_runtime.h>

// ---------------- Problem constants ----------------
#define E_    8
#define H_    2880
#define I_    2880
#define T_    512
#define TOPK_ 2
#define A_    (T_ * TOPK_)    // 1024 assignments
#define N1_   (2 * I_)        // 5760
#define K_    H_              // 2880 (both GEMMs contract over 2880)

#define MTILE   128
#define NTILE   128
#define KSTEP   64
#define LDS_PAD 8
#define MAX_TILES 16          // sum_e ceil(c_e/128) <= 8 + 1024/128 = 16

typedef __bf16 bf16_t;
typedef __bf16 v2bf  __attribute__((ext_vector_type(2)));
typedef __bf16 v8bf  __attribute__((ext_vector_type(8)));
typedef __bf16 v16bf __attribute__((ext_vector_type(16)));
typedef float  v2f   __attribute__((ext_vector_type(2)));
typedef float  v8f   __attribute__((ext_vector_type(8)));

#if __has_builtin(__builtin_amdgcn_cvt_pk_fp8_f32) && __has_builtin(__builtin_amdgcn_cvt_pk_f32_fp8)
#define USE_HW_FP8 1
#endif
#if __has_builtin(__builtin_amdgcn_cvt_pk_bf16_f32)
#define USE_HW_PKBF16 1
#endif

__device__ __forceinline__ v2bf pack_bf16(float a, float b) {
#ifdef USE_HW_PKBF16
  return __builtin_amdgcn_cvt_pk_bf16_f32(a, b);
#else
  v2bf r; r[0] = (bf16_t)a; r[1] = (bf16_t)b; return r;
#endif
}

#ifndef USE_HW_FP8
// fp8 e4m3 quant-dequant (RNE), |w| <= 448 (software fallback)
__device__ __forceinline__ float e4m3_qd(float w) {
  unsigned u = __float_as_uint(w);
  unsigned aexp = (u >> 23) & 0xFFu;
  if (aexp >= 121u) {                 // |w| >= 2^-6 : normal e4m3, keep 3 mantissa bits RNE
    unsigned lsb = (u >> 20) & 1u;
    u += 0x0007FFFFu + lsb;
    u &= 0xFFF00000u;
    return __uint_as_float(u);
  }
  return rintf(w * 512.0f) * (1.0f / 512.0f);   // e4m3 subnormal grid 2^-9
}
#endif

// MXFP8 block QDQ: 32 contiguous floats -> bf16 (dequantized fp8 is exact in bf16)
__device__ __forceinline__ void qdq_block32(const float* __restrict__ src, bf16_t* dst) {
  float v[32];
  const float4* s4 = reinterpret_cast<const float4*>(src);
#pragma unroll
  for (int i = 0; i < 8; ++i) {
    float4 f = s4[i];
    v[4 * i + 0] = f.x; v[4 * i + 1] = f.y; v[4 * i + 2] = f.z; v[4 * i + 3] = f.w;
  }
  float m = 0.0f;
#pragma unroll
  for (int i = 0; i < 32; ++i) m = fmaxf(m, fabsf(v[i]));
  float ds = m / 448.0f;                                          // maxv / MAX_FP8
  unsigned di = (__float_as_uint(ds) + 0x7FFFFFu) & 0x7F800000u;  // UE8M0 round-up
  float s = __uint_as_float(di);
  if (s == 0.0f) s = 1.1754944e-38f;
  float inv = 1.0f / s;  // power of two -> x*inv exact and |x*inv| <= 448 by construction
#pragma unroll
  for (int i = 0; i < 4; ++i) {
    v8bf o;
#pragma unroll
    for (int p = 0; p < 4; ++p) {
      float a = v[8 * i + 2 * p] * inv;
      float b = v[8 * i + 2 * p + 1] * inv;
#ifdef USE_HW_FP8
      int packed = __builtin_amdgcn_cvt_pk_fp8_f32(a, b, 0, false);  // RNE e4m3 pack
      v2f d = __builtin_amdgcn_cvt_pk_f32_fp8(packed, false);        // exact unpack
      v2bf pb = pack_bf16(d[0] * s, d[1] * s);
#else
      v2bf pb = pack_bf16(e4m3_qd(a) * s, e4m3_qd(b) * s);
#endif
      o[2 * p]     = pb[0];
      o[2 * p + 1] = pb[1];
    }
    *reinterpret_cast<v8bf*>(dst + 8 * i) = o;
  }
}

// ---------------- WMMA fragment load from LDS ----------------
// 16-bit A/B frag: lanes 0-15 hold K {0..7, 16..23}, lanes 16-31 hold K {8..15, 24..31}
__device__ __forceinline__ v16bf frag_ld(const bf16_t* p, int khalf) {
  v8bf lo = *reinterpret_cast<const v8bf*>(p + 8 * khalf);
  v8bf hi = *reinterpret_cast<const v8bf*>(p + 16 + 8 * khalf);
  return __builtin_shufflevector(lo, hi, 0, 1, 2, 3, 4, 5, 6, 7,
                                         8, 9, 10, 11, 12, 13, 14, 15);
}

// ---------------- utility kernels ----------------
__global__ void zero_kernel(float* __restrict__ p, int n) {
  int i = blockIdx.x * blockDim.x + threadIdx.x;
  if (i < n) p[i] = 0.0f;
}

// Deterministic counting-sort routing + tile descriptors
__global__ void route_kernel(const int* __restrict__ router, int* __restrict__ sorted,
                             int4* __restrict__ tiles, int* __restrict__ tileCnt) {
  __shared__ int eid[A_];
  __shared__ int cnt[E_];
  int t = threadIdx.x;   // blockDim == A_ == 1024
  eid[t] = router[t];
  if (t < E_) cnt[t] = 0;
  __syncthreads();
  atomicAdd(&cnt[eid[t]], 1);   // order-independent counts
  __syncthreads();
  int e = eid[t];
  int pos = 0;
  for (int i = 0; i < A_; ++i) {      // deterministic stable position
    int ei = eid[i];
    pos += (ei < e) ? 1 : 0;
    pos += (i < t && ei == e) ? 1 : 0;
  }
  sorted[pos] = t;                    // assignment id a = token*2 + k
  if (t == 0) {
    int nt = 0, run = 0;
    for (int ee = 0; ee < E_; ++ee) {
      int c = cnt[ee];
      for (int off = 0; off < c; off += MTILE) {
        int4 td;
        td.x = ee; td.y = run + off; td.z = min(MTILE, c - off); td.w = 0;
        tiles[nt++] = td;
      }
      run += c;
    }
    *tileCnt = nt;
  }
}

// ---------------- grouped GEMM core (bf16 WMMA 16x16x32) ----------------
template <bool FIRST>
__global__ __launch_bounds__(256)
void gemm_kernel(const float* __restrict__ hidden,    // FIRST: activation source
                 const float* __restrict__ W,         // [E][NCOLS][K] fp32
                 const float* __restrict__ bias,      // [E][NCOLS]
                 const float* __restrict__ rweights,  // [A] (second gemm)
                 const int*   __restrict__ sorted,    // [A]
                 const int4*  __restrict__ tiles,
                 const int*   __restrict__ tileCnt,
                 const float* __restrict__ interIn,   // second: A source [A][K]
                 float*       __restrict__ interOut,  // first: [A][I]
                 float*       __restrict__ out) {     // second: [T][H]
  const int NCOLS = FIRST ? N1_ : H_;
  if ((int)blockIdx.y >= *tileCnt) return;
  int4 td = tiles[blockIdx.y];
  const int e = td.x, mstart = td.y, msize = td.z;
  const int nbase = blockIdx.x * NTILE;

  __shared__ bf16_t As[MTILE][KSTEP + LDS_PAD];
  __shared__ bf16_t Bs[NTILE][KSTEP + LDS_PAD];

  const int tid  = threadIdx.x;
  const int lane = tid & 31;
  const int wave = tid >> 5;
  const int wm = wave >> 2, wn = wave & 3;   // 2 (M) x 4 (N) waves
  const int lhi = lane >> 4, llo = lane & 15;

  // copy roles: thread handles row/col (tid>>1), 32-elem half (tid&1)
  const int cr = tid >> 1, cb = tid & 1;
  const bool aValid = (cr < msize);
  const float* aSrcRow = nullptr;
  if (FIRST) {
    if (aValid) { int a = sorted[mstart + cr]; aSrcRow = hidden + (size_t)(a >> 1) * K_; }
  } else {
    if (aValid) aSrcRow = interIn + (size_t)(mstart + cr) * K_;
  }
  const int bn = nbase + cr;
  const float* bSrcRow =
      (bn < NCOLS) ? (W + (size_t)e * (size_t)NCOLS * K_ + (size_t)bn * K_) : nullptr;

  // Zero-fill the tail regions ONCE (they are never overwritten inside the loop)
  {
    v8bf z;
#pragma unroll
    for (int j = 0; j < 8; ++j) z[j] = (bf16_t)0.0f;
    if (!aValid) {
#pragma unroll
      for (int i = 0; i < 4; ++i) *reinterpret_cast<v8bf*>(&As[cr][cb * 32 + 8 * i]) = z;
    }
    if (!bSrcRow) {
#pragma unroll
      for (int i = 0; i < 4; ++i) *reinterpret_cast<v8bf*>(&Bs[cr][cb * 32 + 8 * i]) = z;
    }
  }

  v8f acc[4][2];
#pragma unroll
  for (int i = 0; i < 4; ++i)
#pragma unroll
    for (int j = 0; j < 2; ++j)
#pragma unroll
      for (int k = 0; k < 8; ++k) acc[i][j][k] = 0.0f;

  for (int k0 = 0; k0 < K_; k0 += KSTEP) {
    // ---- stage A tile (fused MXFP8 quant-dequant -> bf16) ----
    if (aValid) {
      qdq_block32(aSrcRow + k0 + cb * 32, &As[cr][cb * 32]);
      if (k0 + KSTEP < K_) __builtin_prefetch(aSrcRow + k0 + KSTEP + cb * 32, 0, 1);
    }
    // ---- stage B tile (fp32 weights -> bf16, contiguous-K layout) ----
    if (bSrcRow) {
      const float4* s4 = reinterpret_cast<const float4*>(bSrcRow + k0 + cb * 32);
#pragma unroll
      for (int i = 0; i < 4; ++i) {
        float4 f0 = s4[2 * i], f1 = s4[2 * i + 1];
        v8bf o;
        v2bf p0 = pack_bf16(f0.x, f0.y);
        v2bf p1 = pack_bf16(f0.z, f0.w);
        v2bf p2 = pack_bf16(f1.x, f1.y);
        v2bf p3 = pack_bf16(f1.z, f1.w);
        o[0] = p0[0]; o[1] = p0[1]; o[2] = p1[0]; o[3] = p1[1];
        o[4] = p2[0]; o[5] = p2[1]; o[6] = p3[0]; o[7] = p3[1];
        *reinterpret_cast<v8bf*>(&Bs[cr][cb * 32 + 8 * i]) = o;
      }
      if (k0 + KSTEP < K_) __builtin_prefetch(bSrcRow + k0 + KSTEP + cb * 32, 0, 1);
    }
    __syncthreads();

    // ---- WMMA inner product: 2 k-chunks x (4 M-frags * 2 N-frags) ----
#pragma unroll
    for (int kk = 0; kk < KSTEP; kk += 32) {
      v16bf af[4], bf[2];
#pragma unroll
      for (int i = 0; i < 4; ++i) af[i] = frag_ld(&As[wm * 64 + i * 16 + llo][kk], lhi);
#pragma unroll
      for (int j = 0; j < 2; ++j) bf[j] = frag_ld(&Bs[wn * 32 + j * 16 + llo][kk], lhi);
#pragma unroll
      for (int i = 0; i < 4; ++i)
#pragma unroll
        for (int j = 0; j < 2; ++j)
          acc[i][j] = __builtin_amdgcn_wmma_f32_16x16x32_bf16(
              false, af[i], false, bf[j], (short)0, acc[i][j], false, false);
    }
    __syncthreads();
  }

  // ---- epilogue ----
#pragma unroll
  for (int i = 0; i < 4; ++i) {
#pragma unroll
    for (int j = 0; j < 2; ++j) {
      int ng = nbase + wn * 32 + j * 16 + llo;     // output column
      if (!FIRST && ng >= NCOLS) continue;         // N1_ % 128 == 0: no divergence in FIRST
      float bv = bias[(size_t)e * NCOLS + ng];
#pragma unroll
      for (int jj = 0; jj < 8; ++jj) {
        int r = wm * 64 + i * 16 + jj + 8 * lhi;   // row within tile
        float v = acc[i][j][jj] + bv;
        if (FIRST) {
          // interleaved SwiGLU: even column = gate, odd column = up (lane parity == ng parity)
          float o = __shfl_xor(v, 1, 32);
          if ((lane & 1) == 0 && r < msize) {
            float gate = fminf(v, 7.0f);
            float up   = fminf(fmaxf(o, -7.0f), 7.0f);
            float sig  = 1.0f / (1.0f + __expf(-1.702f * gate));
            interOut[(size_t)(mstart + r) * I_ + (ng >> 1)] = gate * sig * (up + 1.0f);
          }
        } else {
          if (r < msize) {
            int a = sorted[mstart + r];
            float w = rweights[a];
            atomicAdd(&out[(size_t)(a >> 1) * H_ + ng], v * w);
          }
        }
      }
    }
  }
}

// ---------------- launcher ----------------
extern "C" void kernel_launch(void* const* d_in, const int* in_sizes, int n_in,
                              void* d_out, int out_size, void* d_ws, size_t ws_size,
                              hipStream_t stream) {
  (void)in_sizes; (void)n_in; (void)out_size; (void)ws_size;
  const float* hidden = (const float*)d_in[0];
  const int*   router = (const int*)d_in[1];
  const float* rw     = (const float*)d_in[2];
  const float* W1     = (const float*)d_in[3];
  const float* B1     = (const float*)d_in[4];
  const float* W2     = (const float*)d_in[5];
  const float* B2     = (const float*)d_in[6];
  float* out = (float*)d_out;

  char* ws = (char*)d_ws;
  int*   tileCnt = (int*)(ws + 0);
  int4*  tiles   = (int4*)(ws + 16);      // 16 * 16B = 256B
  int*   sorted  = (int*)(ws + 512);      // 1024 * 4B
  float* inter   = (float*)(ws + 8192);   // 1024 * 2880 * 4B = 11.8 MB

  zero_kernel<<<(T_ * H_ + 255) / 256, 256, 0, stream>>>(out, T_ * H_);
  route_kernel<<<1, A_, 0, stream>>>(router, sorted, tiles, tileCnt);
  gemm_kernel<true><<<dim3(N1_ / NTILE, MAX_TILES), 256, 0, stream>>>(
      hidden, W1, B1, nullptr, sorted, tiles, tileCnt, nullptr, inter, nullptr);
  gemm_kernel<false><<<dim3((H_ + NTILE - 1) / NTILE, MAX_TILES), 256, 0, stream>>>(
      nullptr, W2, B2, rw, sorted, tiles, tileCnt, inter, nullptr, out);
}
